// NanoBaseline_87514253623490
// MI455X (gfx1250) — matrix operand order repaired and checked by reading, not compiled, not run on hardware
//
#include <hip/hip_runtime.h>

// ---------------------------------------------------------------------------
// GPT-2-ish forward for MI455X (gfx1250), bf16 WMMA everywhere.
//   B=8 T=1024 D=512 L=6 H=8 DH=64 DFF=2048 V=32000
// GEMMs: 128x128 block tile, async global->LDS staging (ASYNCcnt),
//        8 v_wmma_f32_16x16x32_bf16 per wave per K-step.
// ---------------------------------------------------------------------------

#define Bsz   8
#define Tsz   1024
#define Dsz   512
#define Lsz   6
#define Hsz   8
#define DHsz  64
#define DFFsz 2048
#define Vsz   32000
#define Msz   (Bsz * Tsz)   // 8192 rows

typedef __bf16        v16bf __attribute__((ext_vector_type(16)));
typedef float         f32x8 __attribute__((ext_vector_type(8)));
typedef unsigned int  u32x4 __attribute__((ext_vector_type(4)));
typedef unsigned short u16;

enum : int { F_BIAS = 1, F_GELU = 2, F_RESID = 4, F_OUTBF = 8 };

__device__ __forceinline__ u16 f2bf(float f) {
    unsigned u = __float_as_uint(f);
    u += 0x7FFFu + ((u >> 16) & 1u);        // round-to-nearest-even
    return (u16)(u >> 16);
}

__device__ __forceinline__ f32x8 zero8() {
    f32x8 z = {0.f, 0.f, 0.f, 0.f, 0.f, 0.f, 0.f, 0.f};
    return z;
}

union FragAB { u32x4 q[2]; v16bf v; };

// A-fragment 16x32 bf16: lane -> row (lane&15); lanes<16: K 0-7 & 16-23,
// lanes>=16: K 8-15 & 24-31.  Two 16B loads from row-major [16][>=32].
__device__ __forceinline__ v16bf frag_a(const u16* row0, int ld, int lane) {
    const int m  = lane & 15;
    const int kb = (lane & 16) ? 8 : 0;
    const u16* p = row0 + (size_t)m * ld + kb;
    FragAB f;
    f.q[0] = *(const u32x4*)(p);
    f.q[1] = *(const u32x4*)(p + 16);
    return f.v;
}

// B-fragment 32x16 bf16 (stored as B^T row-major [n][K]): lane -> col (lane&15);
// lanes<16: K 0-7 & 16-23, lanes>=16: K 8-15 & 24-31.  Two 16B loads per lane.
__device__ __forceinline__ v16bf frag_b(const u16* row0, int ld, int lane) {
    const int n  = lane & 15;
    const int kb = (lane & 16) ? 8 : 0;
    const u16* p = row0 + (size_t)n * ld + kb;
    FragAB f;
    f.q[0] = *(const u32x4*)(p);
    f.q[1] = *(const u32x4*)(p + 16);
    return f.v;
}

__device__ __forceinline__ f32x8 wmma_bf16(v16bf a, v16bf b, f32x8 c) {
    return __builtin_amdgcn_wmma_f32_16x16x32_bf16(
        /*neg_a=*/false, a, /*neg_b=*/false, b,
        /*c_mod=*/(short)0, c, /*reuse_a=*/false, /*reuse_b=*/false);
}

// ---- CDNA5 async global->LDS copy (VGPR-bypassing, tracked by ASYNCcnt) ----
__device__ __forceinline__ void async_ld_b128(unsigned lds_off, const void* gaddr) {
    asm volatile("global_load_async_to_lds_b128 %0, %1, off"
                 :: "v"(lds_off), "v"(gaddr)
                 : "memory");
}
__device__ __forceinline__ void wait_async0() {
    asm volatile("s_wait_asynccnt 0x0" ::: "memory");
}

// ---------------------------------------------------------------------------
// GEMM: C[M,N] = A[M,K](bf16) * Bt[N,K]^T(bf16) (+bias, gelu, fp32 residual)
// Block tile 128x128, BK=32, 256 threads = 8 waves (4x2), wave tile 32x64.
// Double-buffered LDS filled by global_load_async_to_lds_b128.
// ---------------------------------------------------------------------------
template <int FLAGS>
__global__ void __launch_bounds__(256)
gemm_kernel(const u16* __restrict__ A, int lda,
            const u16* __restrict__ Bt, int ldb,
            const float* __restrict__ bias,
            const float* resid,
            float* outF, u16* outB,
            int M, int N, int K)
{
    __shared__ __align__(16) u16 sA[2][128][32];
    __shared__ __align__(16) u16 sB[2][128][32];

    const int tid  = threadIdx.x;
    const int lane = tid & 31;
    const int wid  = tid >> 5;
    const int bm   = blockIdx.y * 128;
    const int bn   = blockIdx.x * 128;
    const int wm   = (wid & 3) * 32;        // 4 wave rows
    const int wn   = (wid >> 2) * 64;       // 2 wave cols

    // staging: each thread owns 32B of each tile: row tid>>1, col (tid&1)*16
    const int srow = tid >> 1, scol = (tid & 1) * 16;
    const u16* gA = A  + (size_t)(bm + srow) * lda + scol;
    const u16* gB = Bt + (size_t)(bn + srow) * ldb + scol;

    auto stage = [&](int buf, int koff) {
        const u16* pA = gA + koff;
        const u16* pB = gB + koff;
        async_ld_b128((unsigned)(size_t)(void*)&sA[buf][srow][scol],     pA);
        async_ld_b128((unsigned)(size_t)(void*)&sA[buf][srow][scol + 8], pA + 8);
        async_ld_b128((unsigned)(size_t)(void*)&sB[buf][srow][scol],     pB);
        async_ld_b128((unsigned)(size_t)(void*)&sB[buf][srow][scol + 8], pB + 8);
    };

    f32x8 c[2][4];
#pragma unroll
    for (int i = 0; i < 2; ++i)
#pragma unroll
        for (int j = 0; j < 4; ++j) c[i][j] = zero8();

    stage(0, 0);
    wait_async0();
    __syncthreads();

    const int KT = K >> 5;
    for (int kt = 0; kt < KT; ++kt) {
        const int  cur  = kt & 1;
        const bool more = (kt + 1) < KT;
        if (more) stage(cur ^ 1, (kt + 1) << 5);

        const v16bf a0 = frag_a(&sA[cur][wm][0],      32, lane);
        const v16bf a1 = frag_a(&sA[cur][wm + 16][0], 32, lane);
#pragma unroll
        for (int ni = 0; ni < 4; ++ni) {
            const v16bf b = frag_b(&sB[cur][wn + ni * 16][0], 32, lane);
            c[0][ni] = wmma_bf16(a0, b, c[0][ni]);
            c[1][ni] = wmma_bf16(a1, b, c[1][ni]);
        }
        if (more) wait_async0();   // my async stores to LDS[nxt] have landed
        __syncthreads();           // everyone's have landed
    }

    // epilogue: lane holds col = wn+ni*16+(lane&15), rows vgpr j + 8*(lane>=16)
    const int ln   = lane & 15;
    const int half = (lane & 16) ? 8 : 0;
#pragma unroll
    for (int mi = 0; mi < 2; ++mi)
#pragma unroll
        for (int ni = 0; ni < 4; ++ni) {
            const int col = bn + wn + ni * 16 + ln;
            float bv = 0.f;
            if (FLAGS & F_BIAS) bv = bias[col];
#pragma unroll
            for (int j = 0; j < 8; ++j) {
                const int row = bm + wm + mi * 16 + half + j;
                float v = c[mi][ni][j] + bv;
                if (FLAGS & F_GELU)
                    v = 0.5f * v * (1.0f + erff(v * 0.70710678118654752f));
                if (FLAGS & F_RESID) v += resid[(size_t)row * N + col];
                if (FLAGS & F_OUTBF) outB[(size_t)row * N + col] = f2bf(v);
                else                 outF[(size_t)row * N + col] = v;
            }
        }
}

// ---------------------------------------------------------------------------
// Flash attention: one 16-query tile per wave per (b,h).  Per 32-key chunk:
// 4 WMMAs for S (K=DH=64), online softmax, P via LDS transpose, 4 WMMAs for O.
// ---------------------------------------------------------------------------
__global__ void __launch_bounds__(128)
attn_kernel(const u16* __restrict__ qkv,   // [B*T, 3D] bf16
            const u16* __restrict__ vT,    // [B*H*DH, T] bf16
            u16* __restrict__ y)           // [B*T, D] bf16
{
    __shared__ __align__(16) u16 sP[4][16][32];   // per-wave P scratch

    const int lane = threadIdx.x & 31;
    const int wid  = threadIdx.x >> 5;
    const int qt   = blockIdx.x * 4 + wid;        // query tile in [0, T/16)
    const int bh   = blockIdx.y;
    const int b    = bh >> 3, h = bh & 7;
    const int q0   = qt << 4;
    const int ln   = lane & 15;
    const int half = (lane & 16) ? 8 : 0;

    const u16* qbase = qkv + (size_t)(b * Tsz + q0) * (3 * Dsz) + h * DHsz;
    const u16* kbase = qkv + (size_t)(b * Tsz) * (3 * Dsz) + Dsz + h * DHsz;
    const u16* vbase = vT + (size_t)bh * DHsz * Tsz;

    const v16bf aq0 = frag_a(qbase,      3 * Dsz, lane);   // dh 0..31
    const v16bf aq1 = frag_a(qbase + 32, 3 * Dsz, lane);   // dh 32..63

    f32x8 o0 = zero8(), o1 = zero8(), o2 = zero8(), o3 = zero8();
    float mrow[8], lrow[8];
#pragma unroll
    for (int j = 0; j < 8; ++j) { mrow[j] = -1e30f; lrow[j] = 0.f; }

    const int nch = (q0 + 16 + 31) >> 5;    // 32-key chunks covering keys<=q0+15
    for (int ch = 0; ch < nch; ++ch) {
        const int k0 = ch << 5;
        f32x8 s0 = zero8(), s1 = zero8();
        {
            const u16* kr  = kbase + (size_t)k0 * (3 * Dsz);
            const u16* kr2 = kbase + (size_t)(k0 + 16) * (3 * Dsz);
            s0 = wmma_bf16(aq0, frag_b(kr,       3 * Dsz, lane), s0);
            s0 = wmma_bf16(aq1, frag_b(kr  + 32, 3 * Dsz, lane), s0);
            s1 = wmma_bf16(aq0, frag_b(kr2,      3 * Dsz, lane), s1);
            s1 = wmma_bf16(aq1, frag_b(kr2 + 32, 3 * Dsz, lane), s1);
        }
        float fac[8];
#pragma unroll
        for (int j = 0; j < 8; ++j) {
            const int rt = q0 + half + j;                 // absolute query row
            float v0 = s0[j] * 0.125f;                    // 1/sqrt(64)
            float v1 = s1[j] * 0.125f;
            if (k0 + ln > rt)      v0 = -1e30f;           // causal mask
            if (k0 + 16 + ln > rt) v1 = -1e30f;
            float mx = fmaxf(v0, v1);
#pragma unroll
            for (int d = 1; d < 16; d <<= 1) mx = fmaxf(mx, __shfl_xor(mx, d, 32));
            const float mnew = fmaxf(mrow[j], mx);
            const float f    = __expf(mrow[j] - mnew);
            const float p0   = __expf(v0 - mnew);
            const float p1   = __expf(v1 - mnew);
            float ps = p0 + p1;
#pragma unroll
            for (int d = 1; d < 16; d <<= 1) ps += __shfl_xor(ps, d, 32);
            lrow[j] = lrow[j] * f + ps;
            mrow[j] = mnew;
            fac[j]  = f;
            sP[wid][half + j][ln]      = f2bf(p0);        // transpose via LDS
            sP[wid][half + j][16 + ln] = f2bf(p1);
        }
#pragma unroll
        for (int j = 0; j < 8; ++j) {
            o0[j] *= fac[j]; o1[j] *= fac[j]; o2[j] *= fac[j]; o3[j] *= fac[j];
        }
        __threadfence_block();                            // LDS store->load order
        const v16bf ap = frag_a(&sP[wid][0][0], 32, lane);
        o0 = wmma_bf16(ap, frag_b(vbase + k0,            Tsz, lane), o0);
        o1 = wmma_bf16(ap, frag_b(vbase + 16 * Tsz + k0, Tsz, lane), o1);
        o2 = wmma_bf16(ap, frag_b(vbase + 32 * Tsz + k0, Tsz, lane), o2);
        o3 = wmma_bf16(ap, frag_b(vbase + 48 * Tsz + k0, Tsz, lane), o3);
        __threadfence_block();                            // WAR before next chunk
    }

    u16* yb = y + (size_t)(b * Tsz + q0) * Dsz + h * DHsz;
#pragma unroll
    for (int j = 0; j < 8; ++j) {
        const int r = half + j;
        const float inv = 1.0f / lrow[j];
        yb[(size_t)r * Dsz + ln]      = f2bf(o0[j] * inv);
        yb[(size_t)r * Dsz + 16 + ln] = f2bf(o1[j] * inv);
        yb[(size_t)r * Dsz + 32 + ln] = f2bf(o2[j] * inv);
        yb[(size_t)r * Dsz + 48 + ln] = f2bf(o3[j] * inv);
    }
}

// ---------------------------------------------------------------------------
// Elementwise / prep kernels
// ---------------------------------------------------------------------------
__global__ void __launch_bounds__(256)
embed_kernel(const int* __restrict__ idx, const float* __restrict__ wte,
             const float* __restrict__ wpe, float* __restrict__ x)
{
    const int row = blockIdx.x, tid = threadIdx.x;
    const int t = row & (Tsz - 1);
    const int v = idx[row];
    x[(size_t)row * Dsz + tid]       = wte[(size_t)v * Dsz + tid]       + wpe[(size_t)t * Dsz + tid];
    x[(size_t)row * Dsz + tid + 256] = wte[(size_t)v * Dsz + tid + 256] + wpe[(size_t)t * Dsz + tid + 256];
}

__global__ void __launch_bounds__(256)
ln_kernel(const float* __restrict__ x, const float* __restrict__ g,
          const float* __restrict__ be, u16* __restrict__ out)
{
    __shared__ float red[256];
    const int row = blockIdx.x, tid = threadIdx.x;
    const float* xr = x + (size_t)row * Dsz;
    const float v0 = xr[tid], v1 = xr[tid + 256];
    red[tid] = v0 + v1;
    __syncthreads();
    for (int d = 128; d > 0; d >>= 1) {
        if (tid < d) red[tid] += red[tid + d];
        __syncthreads();
    }
    const float mean = red[0] * (1.0f / Dsz);
    __syncthreads();
    const float d0 = v0 - mean, d1 = v1 - mean;
    red[tid] = d0 * d0 + d1 * d1;
    __syncthreads();
    for (int d = 128; d > 0; d >>= 1) {
        if (tid < d) red[tid] += red[tid + d];
        __syncthreads();
    }
    const float rstd = rsqrtf(red[0] * (1.0f / Dsz) + 1e-5f);
    out[(size_t)row * Dsz + tid]       = f2bf(d0 * rstd * g[tid] + be[tid]);
    out[(size_t)row * Dsz + tid + 256] = f2bf(d1 * rstd * g[tid + 256] + be[tid + 256]);
}

__global__ void __launch_bounds__(256)
cvt_kernel(const float* __restrict__ in, u16* __restrict__ out, int n)
{
    const int i = blockIdx.x * 256 + threadIdx.x;
    if (i < n) out[i] = f2bf(in[i]);
}

// in[K][N] fp32  ->  out[N][K] bf16
__global__ void __launch_bounds__(256)
cvtT_kernel(const float* __restrict__ in, u16* __restrict__ out, int K, int N)
{
    const int i = blockIdx.x * 256 + threadIdx.x;
    if (i < K * N) {
        const int nn = i / K, kk = i - nn * K;
        out[i] = f2bf(in[(size_t)kk * N + nn]);
    }
}

// qkv[B*T,3D] -> vT[B*H*DH, T]
__global__ void __launch_bounds__(256)
vtrans_kernel(const u16* __restrict__ qkv, u16* __restrict__ vT)
{
    const int i  = blockIdx.x * 256 + threadIdx.x;     // B*H*DH*T total (pow2)
    const int t  = i & (Tsz - 1);
    const int dh = (i >> 10) & (DHsz - 1);
    const int bh = i >> 16;
    const int b = bh >> 3, h = bh & 7;
    vT[i] = qkv[(size_t)(b * Tsz + t) * (3 * Dsz) + 2 * Dsz + h * DHsz + dh];
}

// ---------------------------------------------------------------------------
// Host driver
// ---------------------------------------------------------------------------
extern "C" void kernel_launch(void* const* d_in, const int* in_sizes, int n_in,
                              void* d_out, int out_size, void* d_ws, size_t ws_size,
                              hipStream_t stream)
{
    const int*   idx   = (const int*)  d_in[0];
    const float* wte   = (const float*)d_in[1];
    const float* wpe   = (const float*)d_in[2];
    const float* ln1_g = (const float*)d_in[3];
    const float* ln1_b = (const float*)d_in[4];
    const float* w_qkv = (const float*)d_in[5];
    const float* b_qkv = (const float*)d_in[6];
    const float* w_ao  = (const float*)d_in[7];
    const float* b_ao  = (const float*)d_in[8];
    const float* ln2_g = (const float*)d_in[9];
    const float* ln2_b = (const float*)d_in[10];
    const float* w_fc  = (const float*)d_in[11];
    const float* b_fc  = (const float*)d_in[12];
    const float* w_pr  = (const float*)d_in[13];
    const float* b_pr  = (const float*)d_in[14];
    const float* lnf_g = (const float*)d_in[15];
    const float* lnf_b = (const float*)d_in[16];

    char* p = (char*)d_ws;
    auto wsalloc = [&](size_t bytes) -> char* {
        char* r = p;
        p += (bytes + 255) & ~(size_t)255;
        return r;
    };
    u16*   wteB  = (u16*)  wsalloc((size_t)Vsz * Dsz * 2);
    u16*   wqkvT = (u16*)  wsalloc((size_t)Lsz * 3 * Dsz * Dsz * 2);
    u16*   waoT  = (u16*)  wsalloc((size_t)Lsz * Dsz * Dsz * 2);
    u16*   wfcT  = (u16*)  wsalloc((size_t)Lsz * DFFsz * Dsz * 2);
    u16*   wprT  = (u16*)  wsalloc((size_t)Lsz * Dsz * DFFsz * 2);
    float* x     = (float*)wsalloc((size_t)Msz * Dsz * 4);
    u16*   hB    = (u16*)  wsalloc((size_t)Msz * Dsz * 2);
    u16*   qkvB  = (u16*)  wsalloc((size_t)Msz * 3 * Dsz * 2);
    u16*   vTB   = (u16*)  wsalloc((size_t)Bsz * Hsz * DHsz * Tsz * 2);
    u16*   yB    = (u16*)  wsalloc((size_t)Msz * Dsz * 2);
    u16*   gB    = (u16*)  wsalloc((size_t)Msz * DFFsz * 2);

    // ---- weight precision prep (bf16, pre-transposed to [N][K]) ----
    cvt_kernel<<<(Vsz * Dsz + 255) / 256, 256, 0, stream>>>(wte, wteB, Vsz * Dsz);
    for (int l = 0; l < Lsz; ++l) {
        cvtT_kernel<<<(Dsz * 3 * Dsz + 255) / 256, 256, 0, stream>>>(
            w_qkv + (size_t)l * Dsz * 3 * Dsz, wqkvT + (size_t)l * 3 * Dsz * Dsz, Dsz, 3 * Dsz);
        cvtT_kernel<<<(Dsz * Dsz + 255) / 256, 256, 0, stream>>>(
            w_ao + (size_t)l * Dsz * Dsz, waoT + (size_t)l * Dsz * Dsz, Dsz, Dsz);
        cvtT_kernel<<<(Dsz * DFFsz + 255) / 256, 256, 0, stream>>>(
            w_fc + (size_t)l * Dsz * DFFsz, wfcT + (size_t)l * DFFsz * Dsz, Dsz, DFFsz);
        cvtT_kernel<<<(DFFsz * Dsz + 255) / 256, 256, 0, stream>>>(
            w_pr + (size_t)l * DFFsz * Dsz, wprT + (size_t)l * Dsz * DFFsz, DFFsz, Dsz);
    }

    // ---- forward ----
    embed_kernel<<<Msz, 256, 0, stream>>>(idx, wte, wpe, x);

    constexpr int FL_QKV = F_BIAS | F_OUTBF;
    constexpr int FL_AO  = F_BIAS | F_RESID;
    constexpr int FL_FC  = F_BIAS | F_GELU | F_OUTBF;
    constexpr int FL_PR  = F_BIAS | F_RESID;

    for (int l = 0; l < Lsz; ++l) {
        ln_kernel<<<Msz, 256, 0, stream>>>(x, ln1_g + l * Dsz, ln1_b + l * Dsz, hB);
        gemm_kernel<FL_QKV><<<dim3(3 * Dsz / 128, Msz / 128), 256, 0, stream>>>(
            hB, Dsz, wqkvT + (size_t)l * 3 * Dsz * Dsz, Dsz,
            b_qkv + (size_t)l * 3 * Dsz, nullptr, nullptr, qkvB, Msz, 3 * Dsz, Dsz);
        vtrans_kernel<<<(Bsz * Hsz * DHsz * Tsz) / 256, 256, 0, stream>>>(qkvB, vTB);
        attn_kernel<<<dim3(Tsz / 64, Bsz * Hsz), 128, 0, stream>>>(qkvB, vTB, yB);
        gemm_kernel<FL_AO><<<dim3(Dsz / 128, Msz / 128), 256, 0, stream>>>(
            yB, Dsz, waoT + (size_t)l * Dsz * Dsz, Dsz,
            b_ao + (size_t)l * Dsz, x, x, nullptr, Msz, Dsz, Dsz);
        ln_kernel<<<Msz, 256, 0, stream>>>(x, ln2_g + l * Dsz, ln2_b + l * Dsz, hB);
        gemm_kernel<FL_FC><<<dim3(DFFsz / 128, Msz / 128), 256, 0, stream>>>(
            hB, Dsz, wfcT + (size_t)l * DFFsz * Dsz, Dsz,
            b_fc + (size_t)l * DFFsz, nullptr, nullptr, gB, Msz, DFFsz, Dsz);
        gemm_kernel<FL_PR><<<dim3(Dsz / 128, Msz / 128), 256, 0, stream>>>(
            gB, DFFsz, wprT + (size_t)l * Dsz * DFFsz, DFFsz,
            b_pr + (size_t)l * Dsz, x, x, nullptr, Msz, Dsz, DFFsz);
    }

    ln_kernel<<<Msz, 256, 0, stream>>>(x, lnf_g, lnf_b, hB);
    // tied head: Bt = wte (already [V][D] = B^T), fp32 logits straight to d_out
    gemm_kernel<0><<<dim3(Vsz / 128, Msz / 128), 256, 0, stream>>>(
        hB, Dsz, wteB, Dsz, nullptr, nullptr, (float*)d_out, nullptr, Msz, Vsz, Dsz);
}
